// DaRNN_decoder_28226525069978
// MI455X (gfx1250) — matrix operand order repaired and checked by reading, not compiled, not run on hardware
//
#include <hip/hip_runtime.h>
#include <cstdint>

// ---------------------------------------------------------------------------
// DA-RNN decoder for MI455X (gfx1250), fused LDS-resident scan + WMMA GEMMs.
//   B=4096, T=128 -> S=127, P=64, M=64
// Kernel A: enc_proj = input_encoded @ W1_e^T (WMMA f16, f32 accum, f16 out)
// Kernel C: pre-swizzle W1_hc / W_hh into WMMA-B lane layout (f16, in ws)
// Kernel B: 127-step scan, 8 batch rows per workgroup, x & enc_proj in LDS,
//           weight images pulled in with GLOBAL_LOAD_ASYNC_TO_LDS_B128.
// ---------------------------------------------------------------------------

typedef __attribute__((ext_vector_type(16))) _Float16 v16h;
typedef __attribute__((ext_vector_type(8)))  _Float16 v8h;
typedef __attribute__((ext_vector_type(8)))  float    v8f;

constexpr int B_   = 4096;
constexpr int S_   = 127;   // T-1
constexpr int P_   = 64;
constexpr int M_   = 64;
constexpr int ROWS = 8;     // batch rows per workgroup (LDS-resident)

static __device__ __forceinline__ float fast_tanh(float x){
#if __has_builtin(__builtin_amdgcn_tanhf)
  return __builtin_amdgcn_tanhf(x);     // v_tanh_f32 on gfx1250
#else
  return tanhf(x);
#endif
}
static __device__ __forceinline__ float fast_sigmoid(float x){
  return 1.0f / (1.0f + __expf(-x));
}

// issue one 16-byte async global->LDS copy (ASYNCcnt-tracked)
static __device__ __forceinline__ void async_copy_b128(void* ldsDst, const void* gSrc){
  uint32_t d = (uint32_t)(uintptr_t)ldsDst;               // low 32 bits = LDS byte addr
  unsigned long long s = (unsigned long long)(uintptr_t)gSrc;
  asm volatile("global_load_async_to_lds_b128 %0, %1, off"
               :: "v"(d), "v"(s) : "memory");
}
static __device__ __forceinline__ void async_wait_all(){
  asm volatile("s_wait_asynccnt 0x0" ::: "memory");
}

// ------------------------- workspace layout (bytes) -------------------------
constexpr size_t WS_EP   = 0;                                  // f16 enc_proj [B*S, 64]
constexpr size_t WS_W1B  = (size_t)B_ * S_ * P_ * 2;           // f16 W1_hc  B-swizzled [4 qt][4 ck][32][16]
constexpr size_t WS_WHHB = WS_W1B + (size_t)8192 * 2;          // f16 W_hh   B-swizzled [16 nt][2 ck][32][16]
constexpr size_t WS_END  = WS_WHHB + (size_t)16384 * 2;        // ~66.7 MB total

// ------------------------- LDS layout (scan kernel) -------------------------
constexpr size_t align32(size_t x){ return (x + 31) & ~size_t(31); }
constexpr size_t OFF_EP  = 0;                                              // f16 [ROWS][S][65] (padded stride)
constexpr size_t OFF_X   = align32(OFF_EP  + (size_t)ROWS*S_*65*2);        // f16 [ROWS][S][64]
constexpr size_t OFF_WHH = align32(OFF_X   + (size_t)ROWS*S_*P_*2);        // f16 W_hh B image (async-filled)
constexpr size_t OFF_HC  = align32(OFF_WHH + (size_t)16384*2);             // f16 [16][128] hc image (rows 8..15 zero)
constexpr size_t OFF_H   = align32(OFF_HC  + (size_t)16*128*2);            // f32 [ROWS][64]
constexpr size_t OFF_C   = align32(OFF_H   + (size_t)ROWS*M_*4);
constexpr size_t OFF_UW  = align32(OFF_C   + (size_t)ROWS*M_*4);           // f32 [ROWS][64][2] = (u+b1, w2)
constexpr size_t OFF_A   = align32(OFF_UW  + (size_t)ROWS*128*4);          // f32 [ROWS][128] softmax probs
constexpr size_t OFF_CTX = align32(OFF_A   + (size_t)ROWS*128*4);          // f32 [ROWS][64]
constexpr size_t OFF_G   = align32(OFF_CTX + (size_t)ROWS*P_*4);           // f32 [ROWS][256] raw gate GEMM
constexpr size_t OFF_YT  = align32(OFF_G   + (size_t)ROWS*256*4);          // f32 [ROWS]
constexpr size_t OFF_B1  = align32(OFF_YT  + (size_t)ROWS*4);
constexpr size_t OFF_W2  = align32(OFF_B1  + 64*4);
constexpr size_t OFF_WIH = align32(OFF_W2  + 64*4);
constexpr size_t OFF_BIH = align32(OFF_WIH + 256*4);
constexpr size_t OFF_BHH = align32(OFF_BIH + 256*4);
constexpr size_t OFF_FCW = align32(OFF_BHH + 256*4);
constexpr size_t OFF_FCF = align32(OFF_FCW + 65*4);
constexpr size_t OFF_SCA = align32(OFF_FCF + 128*4);                       // fc_b, attn_b2, fcf_b
constexpr size_t SMEM_BYTES = align32(OFF_SCA + 8*4);                      // ~326 KB (<= 327,680 WGP limit)

// ---------------------------------------------------------------------------
// Kernel A: enc_proj[b,s,q] = sum_p x[b,s,p] * attn_W1[q,128+p]   (f16 out)
// ---------------------------------------------------------------------------
__global__ void __launch_bounds__(128)
encproj_wmma(const float* __restrict__ x, const float* __restrict__ W1,
             _Float16* __restrict__ ep)
{
  const int wave = threadIdx.x >> 5;
  const int lane = threadIdx.x & 31;
  const int m    = lane & 15;
  const int hi   = lane >> 4;
  const long row0 = (long)blockIdx.x * 16;      // rows of flattened [B*S, P]
  const float* xr = x + (row0 + m) * P_;        // A row for this lane
  const int   q  = wave * 16 + m;               // output column (N index)
  const float* wq = W1 + q * 192 + 128;         // W1_e slice: attn_W1[q, 128:192]

  v8f acc = {};
#pragma unroll
  for (int ck = 0; ck < 2; ++ck){               // K = 64 -> two K=32 chunks
    const int kb = ck * 32;
    v16h A, Bv;
#pragma unroll
    for (int j = 0; j < 16; ++j){               // ISA 16-bit A layout
      int k = kb + hi*8 + j + ((j >= 8) ? 8 : 0);
      A[j] = (_Float16)xr[k];
    }
#pragma unroll
    for (int e = 0; e < 16; ++e){               // ISA 16-bit B layout
      int k = kb + hi*8 + e + ((e >= 8) ? 8 : 0);
      Bv[e] = (_Float16)wq[k];
    }
    acc = __builtin_amdgcn_wmma_f32_16x16x32_f16(false, A, false, Bv,
                                                 (short)0, acc, false, false);
  }
  _Float16* op = ep + (row0 + hi*8) * P_ + wave*16 + m;   // C: m=r+8*hi, n=lane&15
#pragma unroll
  for (int r = 0; r < 8; ++r)
    op[(long)r * P_] = (_Float16)acc[r];
}

// ---------------------------------------------------------------------------
// Kernel C: one-time weight swizzles into WMMA-B lane layout (f16, ws).
//   w1B [qt(4)][ck(4)][lane(32)][e(16)] : B[k][q] = attn_W1[q, k], k<128
//   whhB[nt(16)][ck(2)][lane(32)][e(16)]: B[k][n] = W_hh[n, k],    k<64
// ---------------------------------------------------------------------------
__global__ void __launch_bounds__(256)
swizzle_weights(const float* __restrict__ W1, const float* __restrict__ Whh,
                _Float16* __restrict__ w1B, _Float16* __restrict__ whhB)
{
  for (int i = threadIdx.x; i < 8192; i += 256){
    int e = i & 15, l = (i >> 4) & 31, ck = (i >> 9) & 3, qt = i >> 11;
    int q = qt*16 + (l & 15);
    int k = ck*32 + (l >> 4)*8 + e + ((e >= 8) ? 8 : 0);
    w1B[i] = (_Float16)W1[q*192 + k];
  }
  for (int i = threadIdx.x; i < 16384; i += 256){
    int e = i & 15, l = (i >> 4) & 31, ck = (i >> 9) & 1, nt = i >> 10;
    int n = nt*16 + (l & 15);
    int k = ck*32 + (l >> 4)*8 + e + ((e >= 8) ? 8 : 0);
    whhB[i] = (_Float16)Whh[n*64 + k];
  }
}

// ---------------------------------------------------------------------------
// Kernel B: fused 127-step scan. 512 blocks x 256 threads (8 wave32).
// ---------------------------------------------------------------------------
__global__ void __launch_bounds__(256)
darnn_scan(const float* __restrict__ x_g, const float* __restrict__ yhist,
           const _Float16* __restrict__ ep_g,
           const _Float16* __restrict__ w1B, const _Float16* __restrict__ whhBws,
           const float* __restrict__ b1_g, const float* __restrict__ w2_g,
           const float* __restrict__ b2_g, const float* __restrict__ Wih_g,
           const float* __restrict__ bih_g, const float* __restrict__ bhh_g,
           const float* __restrict__ fcW_g, const float* __restrict__ fcb_g,
           const float* __restrict__ fcf_g, const float* __restrict__ fcfb_g,
           float* __restrict__ out)
{
  extern __shared__ char smem[];
  _Float16* ep16 = (_Float16*)(smem + OFF_EP);
  _Float16* x16  = (_Float16*)(smem + OFF_X);
  _Float16* whhB = (_Float16*)(smem + OFF_WHH);
  _Float16* hc16 = (_Float16*)(smem + OFF_HC);
  float* hS   = (float*)(smem + OFF_H);
  float* cS   = (float*)(smem + OFF_C);
  float* uwS  = (float*)(smem + OFF_UW);
  float* aS   = (float*)(smem + OFF_A);
  float* ctxS = (float*)(smem + OFF_CTX);
  float* gS   = (float*)(smem + OFF_G);
  float* ytS  = (float*)(smem + OFF_YT);
  float* b1   = (float*)(smem + OFF_B1);
  float* w2   = (float*)(smem + OFF_W2);
  float* Wih  = (float*)(smem + OFF_WIH);
  float* bih  = (float*)(smem + OFF_BIH);
  float* bhh  = (float*)(smem + OFF_BHH);
  float* fcw  = (float*)(smem + OFF_FCW);
  float* fcf  = (float*)(smem + OFF_FCF);
  float* scal = (float*)(smem + OFF_SCA);

  const int tid  = threadIdx.x;
  const int wave = tid >> 5;
  const int lane = tid & 31;
  const long rowBase = (long)blockIdx.x * ROWS;

  // ------------------------------ init ------------------------------------
  // (1) ASYNC: pre-swizzled W_hh B image, 2048 x 16B chunks (layout-preserving)
  for (int i = tid; i < 2048; i += 256)
    async_copy_b128(whhB + i*8, whhBws + i*8);
  // (2) ASYNC: small f32 weight vectors, one aligned 16B chunk per thread
  {
    const float* src; size_t dst;
    if      (tid < 64)  { src = Wih_g + tid*4;        dst = OFF_WIH + (size_t)tid*16; }
    else if (tid < 128) { src = bih_g + (tid-64)*4;   dst = OFF_BIH + (size_t)(tid-64)*16; }
    else if (tid < 192) { src = bhh_g + (tid-128)*4;  dst = OFF_BHH + (size_t)(tid-128)*16; }
    else if (tid < 208) { src = b1_g  + (tid-192)*4;  dst = OFF_B1  + (size_t)(tid-192)*16; }
    else if (tid < 224) { src = w2_g  + (tid-208)*4;  dst = OFF_W2  + (size_t)(tid-208)*16; }
    else                { src = fcf_g + (tid-224)*4;  dst = OFF_FCF + (size_t)(tid-224)*16; }
    async_copy_b128(smem + dst, src);
    if (tid < 16)                                    // fc_W[0..63]
      async_copy_b128(smem + OFF_FCW + (size_t)tid*16, fcW_g + tid*4);
  }
  // (3) VGPR-mediated fills that need conversion / de-banking padding
  {
    const float*    xsrc = x_g  + rowBase * S_ * P_;
    const _Float16* esrc = ep_g + rowBase * S_ * P_;
    for (int i = tid; i < ROWS*S_*P_; i += 256){
      x16[i] = (_Float16)xsrc[i];          // f32 -> f16
      int rs = i >> 6, p = i & 63;
      ep16[rs*65 + p] = esrc[i];           // stride-65 pad kills bank conflicts
    }
  }
  for (int i = tid; i < ROWS*M_; i += 256){ hS[i] = 0.f; cS[i] = 0.f; }
  for (int i = tid; i < 16*128;  i += 256)  hc16[i] = (_Float16)0.f;
  if (tid == 0){
    fcw[64] = fcW_g[64];
    scal[0] = fcb_g[0]; scal[1] = b2_g[0]; scal[2] = fcfb_g[0];
    __builtin_prefetch(w1B, 0, 1);         // global_prefetch_b8: warm B tiles
  }
  async_wait_all();                        // ASYNCcnt == 0 before barrier
  __syncthreads();

  // ------------------------------ scan ------------------------------------
  for (int t = 0; t < S_; ++t){
    // Phase U: u = hc @ W1_hc^T  (waves 0..3, one 16-col tile each, K=128)
    if (wave < 4){
      v8f acc = {};
      const int qn = wave*16 + (lane & 15);
      const _Float16* hp0 = hc16 + (lane & 15)*128 + ((lane >> 4) << 3);
#pragma unroll
      for (int ck = 0; ck < 4; ++ck){
        union { v16h v; v8h h[2]; } A;
        const _Float16* hp = hp0 + ck*32;
        A.h[0] = *(const v8h*)hp;          // K j=0..7
        A.h[1] = *(const v8h*)(hp + 16);   // K j=8..15 (+16 in k-space)
        v16h Bv = *(const v16h*)(w1B + (((wave << 2) + ck) << 9) + (lane << 4));
        acc = __builtin_amdgcn_wmma_f32_16x16x32_f16(false, A.v, false, Bv,
                                                     (short)0, acc, false, false);
      }
      if (lane < 16){                      // only M=0..7 rows are real
        float w2v = w2[qn], b1v = b1[qn];
#pragma unroll
        for (int r = 0; r < 8; ++r){
          uwS[(r << 7) + qn*2    ] = acc[r] + b1v;   // pack (u+b1, w2) pairs
          uwS[(r << 7) + qn*2 + 1] = w2v;
        }
      }
    }
    __syncthreads();

    // Phase S: scores + softmax (wave r handles batch row r)
    {
      const int r = wave;
      const float b2v = scal[1];
      const float2* uw = (const float2*)(uwS + (r << 7));
      float sc[4]; float mx = -1e30f;
#pragma unroll
      for (int i = 0; i < 4; ++i){
        int s = lane + (i << 5);
        float v = -1e30f;
        if (s < S_){
          const _Float16* epp = ep16 + (r*S_ + s)*65;
          float acc = b2v;
#pragma unroll 8
          for (int q = 0; q < 64; ++q){
            float2 p = uw[q];                        // one ds_load_b64 (broadcast)
            acc += fast_tanh((float)epp[q] + p.x) * p.y;
          }
          v = acc;
        }
        sc[i] = v;
        mx = fmaxf(mx, v);
      }
#pragma unroll
      for (int off = 16; off > 0; off >>= 1) mx = fmaxf(mx, __shfl_xor(mx, off, 32));
      float pr[4], sum = 0.f;
#pragma unroll
      for (int i = 0; i < 4; ++i){
        pr[i] = (sc[i] > -1e29f) ? __expf(sc[i] - mx) : 0.f;
        sum += pr[i];
      }
#pragma unroll
      for (int off = 16; off > 0; off >>= 1) sum += __shfl_xor(sum, off, 32);
      float inv = 1.0f / sum;
#pragma unroll
      for (int i = 0; i < 4; ++i){
        int s = lane + (i << 5);
        if (s < S_) aS[(r << 7) + s] = pr[i] * inv;
      }
    }
    __syncthreads();

    // Phase C: context = a @ x   (+ fused y_tilde via wave reduction)
    {
      const int r = wave;
      const _Float16* xr = x16 + r*S_*P_;
      const float*    ar = aS + (r << 7);
      float a1 = 0.f, a2 = 0.f;
      for (int s = 0; s < S_; ++s){
        float av = ar[s];
        a1 += av * (float)xr[s*P_ + lane];
        a2 += av * (float)xr[s*P_ + lane + 32];
      }
      ctxS[r*P_ + lane]      = a1;
      ctxS[r*P_ + lane + 32] = a2;
      float yp = a1*fcw[lane] + a2*fcw[lane + 32];
#pragma unroll
      for (int off = 16; off > 0; off >>= 1) yp += __shfl_xor(yp, off, 32);
      if (lane == 0){
        float yv = yhist[(rowBase + r)*S_ + t];
        ytS[r] = yp + yv*fcw[64] + scal[0];
      }
    }
    __syncthreads();

    // Phase G: raw gates = h @ W_hh^T  (8 waves x 2 N-tiles x 2 K-chunks)
    {
#pragma unroll
      for (int j = 0; j < 2; ++j){
        const int nt = wave*2 + j;
        v8f acc = {};
        const _Float16* hp0 = hc16 + (lane & 15)*128 + ((lane >> 4) << 3);
#pragma unroll
        for (int ck = 0; ck < 2; ++ck){            // h occupies k=0..63 of hc
          union { v16h v; v8h h[2]; } A;
          const _Float16* hp = hp0 + ck*32;
          A.h[0] = *(const v8h*)hp;
          A.h[1] = *(const v8h*)(hp + 16);
          v16h Bv = *(const v16h*)(whhB + ((nt*2 + ck) << 9) + (lane << 4));
          acc = __builtin_amdgcn_wmma_f32_16x16x32_f16(false, A.v, false, Bv,
                                                       (short)0, acc, false, false);
        }
        if (lane < 16){
          const int n = nt*16 + lane;
#pragma unroll
          for (int r = 0; r < 8; ++r) gS[(r << 8) + n] = acc[r];
        }
      }
    }
    __syncthreads();

    // Phase Upd: LSTM cell update (fold in y_tilde*W_ih + biases here)
    {
      const int r = wave;
      const float yt = ytS[r];
#pragma unroll
      for (int half = 0; half < 2; ++half){
        const int m = lane + half*32;
        float gi = gS[(r<<8) + m      ] + yt*Wih[m      ] + bih[m      ] + bhh[m      ];
        float gf = gS[(r<<8) + 64 + m ] + yt*Wih[64 + m ] + bih[64 + m ] + bhh[64 + m ];
        float gg = gS[(r<<8) + 128 + m] + yt*Wih[128 + m] + bih[128 + m] + bhh[128 + m];
        float go = gS[(r<<8) + 192 + m] + yt*Wih[192 + m] + bih[192 + m] + bhh[192 + m];
        float cN = fast_sigmoid(gf)*cS[(r<<6)+m] + fast_sigmoid(gi)*fast_tanh(gg);
        float hN = fast_sigmoid(go)*fast_tanh(cN);
        cS[(r<<6)+m] = cN;
        hS[(r<<6)+m] = hN;
        hc16[(r<<7) + m]      = (_Float16)hN;   // keep f16 hc image for WMMA A
        hc16[(r<<7) + 64 + m] = (_Float16)cN;
      }
    }
    __syncthreads();
  }

  // final: out[b] = [h, context] @ fcf_W^T + fcf_b
  if (tid < ROWS){
    const int r = tid;
    float acc = scal[2];
#pragma unroll 8
    for (int jj = 0; jj < 64; ++jj)
      acc += hS[(r<<6)+jj]*fcf[jj] + ctxS[(r<<6)+jj]*fcf[64+jj];
    out[rowBase + r] = acc;
  }
}

// ---------------------------------------------------------------------------
extern "C" void kernel_launch(void* const* d_in, const int* in_sizes, int n_in,
                              void* d_out, int out_size, void* d_ws, size_t ws_size,
                              hipStream_t stream)
{
  (void)in_sizes; (void)n_in; (void)out_size; (void)ws_size;
  const float* x    = (const float*)d_in[0];   // [B,S,P]
  const float* yh   = (const float*)d_in[1];   // [B,S]
  const float* W1   = (const float*)d_in[2];   // [64,192]
  const float* b1   = (const float*)d_in[3];
  const float* w2   = (const float*)d_in[4];
  const float* b2   = (const float*)d_in[5];
  const float* Wih  = (const float*)d_in[6];
  const float* Whh  = (const float*)d_in[7];
  const float* bih  = (const float*)d_in[8];
  const float* bhh  = (const float*)d_in[9];
  const float* fcW  = (const float*)d_in[10];
  const float* fcb  = (const float*)d_in[11];
  const float* fcf  = (const float*)d_in[12];
  const float* fcfb = (const float*)d_in[13];
  float* out = (float*)d_out;

  char* ws = (char*)d_ws;                       // needs WS_END ~= 66.7 MB
  _Float16* ep    = (_Float16*)(ws + WS_EP);
  _Float16* w1B   = (_Float16*)(ws + WS_W1B);
  _Float16* whhB  = (_Float16*)(ws + WS_WHHB);

  const int rowTiles = (B_ * S_) / 16;          // 32512 blocks of 128 threads
  encproj_wmma<<<rowTiles, 128, 0, stream>>>(x, W1, ep);
  swizzle_weights<<<1, 256, 0, stream>>>(W1, Whh, w1B, whhB);

  darnn_scan<<<B_ / ROWS, 256, SMEM_BYTES, stream>>>(
      x, yh, ep, w1B, whhB, b1, w2, b2, Wih, bih, bhh, fcW, fcb, fcf, fcfb, out);
}
